// DenseEdgeConv_49572512530911
// MI455X (gfx1250) — matrix operand order repaired and checked by reading, not compiled, not run on hardware
//
#include <hip/hip_runtime.h>

typedef __attribute__((ext_vector_type(16))) __bf16 v16bf;
typedef __attribute__((ext_vector_type(8)))  __bf16 v8bf;
typedef __attribute__((ext_vector_type(8)))  float  v8f;

#define NPTS 2048
#define DIN  64
#define KNN  16
#define COUT 128

// ---------------- bf16 helpers ----------------
__device__ __forceinline__ unsigned short f2bf_bits(float f) {
    unsigned u = __float_as_uint(f);
    unsigned r = u + 0x7FFFu + ((u >> 16) & 1u);   // round-to-nearest-even
    return (unsigned short)(r >> 16);
}
__device__ __forceinline__ __bf16 f2bf(float f) {
    unsigned short s = f2bf_bits(f);
    __bf16 b;
    __builtin_memcpy(&b, &s, 2);
    return b;
}

// ---------------- weight transpose + bf16 convert ----------------
// W (kd x 128) row-major fp32  ->  Wt (128 x kd) bf16, contiguous per column
__global__ void prep_weights(const float* __restrict__ W1,
                             const float* __restrict__ W2,
                             const float* __restrict__ W3,
                             unsigned short* __restrict__ W1t,
                             unsigned short* __restrict__ W2t,
                             unsigned short* __restrict__ W3t) {
    int id = blockIdx.x * blockDim.x + threadIdx.x;
    const int S1 = 128 * 192, S2 = 2 * 128 * 192, S3 = 2 * 128 * 192 + 128 * 320;
    const float* src; unsigned short* dst; int kd; int local;
    if (id < S1)      { src = W1; dst = W1t; kd = 192; local = id; }
    else if (id < S2) { src = W2; dst = W2t; kd = 192; local = id - S1; }
    else if (id < S3) { src = W3; dst = W3t; kd = 320; local = id - S2; }
    else return;
    int c = local / kd, k = local % kd;
    dst[local] = f2bf_bits(src[(size_t)k * COUT + c]);
}

// ---------------- brute-force KNN (top-16, excl. self) ----------------
// one wave32 per point; 8 points per 256-thread block
__global__ void __launch_bounds__(256) knn_kernel(const float* __restrict__ pos,
                                                  int* __restrict__ idx_out) {
    __shared__ unsigned long long keys[8 * 32 * KNN];
    int tid  = threadIdx.x;
    int wave = tid >> 5, lane = tid & 31;
    int p = blockIdx.x * 8 + wave;            // global point id (< 8192)
    int b = p >> 11, n = p & (NPTS - 1);
    const float* pb = pos + (size_t)b * NPTS * 3;
    float px = pb[n * 3 + 0], py = pb[n * 3 + 1], pz = pb[n * 3 + 2];

    float dist[KNN]; int idx[KNN];
#pragma unroll
    for (int t = 0; t < KNN; ++t) { dist[t] = 3.4e38f; idx[t] = 0x7fffffff; }

    for (int j = lane; j < NPTS; j += 32) {
        if (j == n) continue;
        float dx = pb[j * 3 + 0] - px;
        float dy = pb[j * 3 + 1] - py;
        float dz = pb[j * 3 + 2] - pz;
        float d = dx * dx + dy * dy + dz * dz;
        float cd = d; int ci = j;
#pragma unroll
        for (int t = 0; t < KNN; ++t) {        // sorted insertion (bubble-through)
            if (cd < dist[t]) {
                float td = dist[t]; int ti = idx[t];
                dist[t] = cd; idx[t] = ci; cd = td; ci = ti;
            }
        }
    }

    unsigned long long* my = keys + ((size_t)(wave * 32 + lane)) * KNN;
#pragma unroll
    for (int t = 0; t < KNN; ++t)
        my[t] = ((unsigned long long)__float_as_uint(dist[t]) << 32) | (unsigned)idx[t];
    __syncthreads();

    int h = 0;
    for (int s = 0; s < KNN; ++s) {
        unsigned long long k = (h < KNN) ? my[h] : ~0ull;
        unsigned long long r = k;
#pragma unroll
        for (int off = 16; off >= 1; off >>= 1) {
            unsigned long long o = __shfl_xor(r, off, 32);
            r = (o < r) ? o : r;
        }
        if (k == r) ++h;                       // winner advances (keys unique)
        if (lane == 0) idx_out[(size_t)p * KNN + s] = (int)(r & 0xffffffffu);
    }
}

// ---------------- one 16x16 WMMA GEMM tile (K = KSTEPS*32) ----------------
template <int KSTEPS>
__device__ __forceinline__ v8f gemm16(const __bf16* __restrict__ Arow,   // LDS row ptr (row m = lane&15)
                                      const __bf16* __restrict__ Wrow,   // global bf16 col ptr (col c)
                                      int h) {
    v8f acc = {0.f, 0.f, 0.f, 0.f, 0.f, 0.f, 0.f, 0.f};
#pragma unroll
    for (int ks = 0; ks < KSTEPS; ++ks) {
        const __bf16* a0 = Arow + ks * 32 + h * 16;
        v8bf alo = *(const v8bf*)(a0);          // K = ks*32 + h*16 .. +7
        v8bf ahi = *(const v8bf*)(a0 + 8);      // K = ks*32 + h*16 + 8 .. +15
        v16bf av = __builtin_shufflevector(alo, ahi,
                        0, 1, 2, 3, 4, 5, 6, 7, 8, 9, 10, 11, 12, 13, 14, 15);
        v16bf bv = *(const v16bf*)(Wrow + ks * 32 + h * 16);   // K = ks*32 + h*16 .. +15
        acc = __builtin_amdgcn_wmma_f32_16x16x32_bf16(
                  false, av, false, bv, (short)0, acc, false, false);
    }
    return acc;
}

// ---------------- fused edge-MLP + max-over-K ----------------
// one block per point; 8 waves, each owns 16 output channels
__global__ void __launch_bounds__(256)
fused_edgeconv(const float* __restrict__ x,
               const float* __restrict__ b1,
               const float* __restrict__ b2,
               const float* __restrict__ b3,
               const unsigned short* __restrict__ W1t,   // bf16 [128][192]
               const unsigned short* __restrict__ W2t,   // bf16 [128][192]
               const unsigned short* __restrict__ W3t,   // bf16 [128][320]
               const int* __restrict__ knn,
               float* __restrict__ out) {
    __shared__ __attribute__((aligned(16))) __bf16 A1[16 * 192];  // edge features
    __shared__ __attribute__((aligned(16))) __bf16 A3[16 * 320];  // [y2 | y1 | xc]

    int tid  = threadIdx.x;
    int wave = tid >> 5, lane = tid & 31;
    int p = blockIdx.x;
    int b = p >> 11, n = p & (NPTS - 1);
    const float* xb = x + (size_t)b * NPTS * DIN;
    const float* xc = xb + (size_t)n * DIN;
    float* op = out + (size_t)p * 448;

    // ---- build edge matrix (16 x 192) and xc tail of A3, copy x -> out tail
    {
        int kk = tid >> 4, sub = tid & 15;
        int nb = knn[(size_t)p * KNN + kk];
        const float* xn = xb + (size_t)nb * DIN;
#pragma unroll
        for (int q = 0; q < 12; ++q) {
            int c = sub * 12 + q;
            float v;
            if (c < 64)       v = xc[c];
            else if (c < 128) v = xn[c - 64];
            else              v = xn[c - 128] - xc[c - 128];
            A1[kk * 192 + c] = f2bf(v);
        }
#pragma unroll
        for (int q = 0; q < 4; ++q) {
            int i = sub * 4 + q;
            A3[kk * 320 + 256 + i] = f2bf(xc[i]);   // x_tiled part of concat
        }
        if (tid < 64) op[384 + tid] = xc[tid];      // max over identical rows = xc
    }
    __syncthreads();

    int h  = lane >> 4;          // K-half select (A) / K-block select (B)
    int nl = lane & 15;          // A: row m ; B/D: column n
    int c  = wave * 16 + nl;     // output channel owned by this lane

    // ---- stage 1: y1 = relu(edge @ W1 + b1)  (K = 192)
    {
        v8f acc = gemm16<6>(A1 + nl * 192, (const __bf16*)W1t + (size_t)c * 192, h);
        float bias = b1[c];
        float mx = -3.4e38f;
#pragma unroll
        for (int r = 0; r < 8; ++r) {
            float v = acc[r] + bias;
            v = v > 0.f ? v : 0.f;
            mx = v > mx ? v : mx;
            A3[(r + 8 * h) * 320 + 128 + c] = f2bf(v);   // y1 into concat buffer
        }
        float o = __shfl_xor(mx, 16, 32);
        mx = o > mx ? o : mx;
        if (lane < 16) op[256 + c] = mx;
    }
    __syncthreads();

    // ---- stage 2: y2 = relu([y1|xc] @ W2 + b2)  (K = 192)
    {
        v8f acc = gemm16<6>(A3 + nl * 320 + 128, (const __bf16*)W2t + (size_t)c * 192, h);
        float bias = b2[c];
        float mx = -3.4e38f;
#pragma unroll
        for (int r = 0; r < 8; ++r) {
            float v = acc[r] + bias;
            v = v > 0.f ? v : 0.f;
            mx = v > mx ? v : mx;
            A3[(r + 8 * h) * 320 + c] = f2bf(v);         // y2 into concat buffer
        }
        float o = __shfl_xor(mx, 16, 32);
        mx = o > mx ? o : mx;
        if (lane < 16) op[128 + c] = mx;
    }
    __syncthreads();

    // ---- stage 3: y3 = [y2|y1|xc] @ W3 + b3  (K = 320, no relu)
    {
        v8f acc = gemm16<10>(A3 + nl * 320, (const __bf16*)W3t + (size_t)c * 320, h);
        float bias = b3[c];
        float mx = -3.4e38f;
#pragma unroll
        for (int r = 0; r < 8; ++r) {
            float v = acc[r] + bias;
            mx = v > mx ? v : mx;
        }
        float o = __shfl_xor(mx, 16, 32);
        mx = o > mx ? o : mx;
        if (lane < 16) op[c] = mx;
    }
}

// ---------------- launcher ----------------
extern "C" void kernel_launch(void* const* d_in, const int* in_sizes, int n_in,
                              void* d_out, int out_size, void* d_ws, size_t ws_size,
                              hipStream_t stream) {
    const float* x   = (const float*)d_in[0];
    const float* pos = (const float*)d_in[1];
    const float* W1  = (const float*)d_in[2];
    const float* b1  = (const float*)d_in[3];
    const float* W2  = (const float*)d_in[4];
    const float* b2  = (const float*)d_in[5];
    const float* W3  = (const float*)d_in[6];
    const float* b3  = (const float*)d_in[7];
    float* out = (float*)d_out;

    char* ws = (char*)d_ws;
    unsigned short* W1t = (unsigned short*)(ws);                  // 128*192*2 = 49152 B
    unsigned short* W2t = (unsigned short*)(ws + 49152);          // 49152 B
    unsigned short* W3t = (unsigned short*)(ws + 98304);          // 128*320*2 = 81920 B
    int* knn_idx        = (int*)(ws + 180224);                    // 8192*16*4 = 524288 B

    const int wtotal = 2 * 128 * 192 + 128 * 320;                 // 90112
    prep_weights<<<(wtotal + 255) / 256, 256, 0, stream>>>(W1, W2, W3, W1t, W2t, W3t);

    const int npoints = 4 * NPTS;                                 // 8192
    knn_kernel<<<npoints / 8, 256, 0, stream>>>(pos, knn_idx);

    fused_edgeconv<<<npoints, 256, 0, stream>>>(x, b1, b2, b3, W1t, W2t, W3t, knn_idx, out);
}
